// RuntimeAttention_50861002719384
// MI455X (gfx1250) — matrix operand order repaired and checked by reading, not compiled, not run on hardware
//
#include <hip/hip_runtime.h>
#include <hip/hip_bf16.h>

// ---------- types ----------
typedef __attribute__((ext_vector_type(16))) __bf16   v16bf;
typedef __attribute__((ext_vector_type(8)))  float    v8f;
typedef __attribute__((ext_vector_type(4)))  unsigned u32x4;
typedef __attribute__((ext_vector_type(2)))  unsigned u32x2;
typedef __attribute__((ext_vector_type(4)))  float    f32x4;
typedef __attribute__((ext_vector_type(4)))  int      v4i;

union Frag {                 // one WMMA 16x16x32 bf16 operand (16 bf16 per lane)
    u32x4 u[2];
    v16bf v;
};
union U4S {                  // element access into a 16-byte value
    u32x4 u;
    unsigned short s[8];
};

__device__ __forceinline__ unsigned short f2bf(float f) {
    unsigned u = __builtin_bit_cast(unsigned, f);
    return (unsigned short)((u + 0x7FFFu + ((u >> 16) & 1u)) >> 16);  // RNE
}
__device__ __forceinline__ float bf2f(unsigned short h) {
    unsigned u = ((unsigned)h) << 16;
    return __builtin_bit_cast(float, u);
}

#define WMMA_BF16(A, B, C) \
    __builtin_amdgcn_wmma_f32_16x16x32_bf16(false, (A), false, (B), (short)0, (C), false, false)

// ---------- CDNA5 async global->LDS copy (ASYNCcnt path), with sync fallback ----------
#if defined(__gfx1250__) && __has_builtin(__builtin_amdgcn_global_load_async_to_lds_b128) && \
    __has_builtin(__builtin_amdgcn_s_wait_asynccnt)
#define HAS_ASYNC 1
#else
#define HAS_ASYNC 0
#endif

__device__ __forceinline__ void cp16(void* lds, const void* g) {
#if HAS_ASYNC
    // builtin prototype (per compiler diagnostic): generic pointers to int4
    __builtin_amdgcn_global_load_async_to_lds_b128((v4i*)g, (v4i*)lds, 0, 0);
#else
    *(u32x4*)lds = *(const u32x4*)g;
#endif
}
template <int N>
__device__ __forceinline__ void wait_async() {
#if HAS_ASYNC
    __builtin_amdgcn_s_wait_asynccnt((short)N);
#endif
}

// =====================================================================
// Pre-pass 1: x f32 [16384,1024] -> bf16 (once; removes repeated cvt)
// =====================================================================
__global__ __launch_bounds__(256) void xconv_kernel(const float* __restrict__ x,
                                                    unsigned short* __restrict__ xb) {
    const int idx = blockIdx.x * 256 + threadIdx.x;     // one f32x4 per thread
    const f32x4 f = *(const f32x4*)(x + (size_t)idx * 4);
    u32x2 p;
    p.x = ((unsigned)f2bf(f.y) << 16) | f2bf(f.x);
    p.y = ((unsigned)f2bf(f.w) << 16) | f2bf(f.z);
    *(u32x2*)(xb + (size_t)idx * 4) = p;
}

// =====================================================================
// Pre-pass 2: W f32 [1024 k][1024 n] -> WT bf16 [1024 n][1024 k] (5 mats)
// =====================================================================
__global__ __launch_bounds__(256) void wconvT_kernel(
    const float* __restrict__ w0, const float* __restrict__ w1, const float* __restrict__ w2,
    const float* __restrict__ w3, const float* __restrict__ w4,
    unsigned short* __restrict__ t0, unsigned short* __restrict__ t1, unsigned short* __restrict__ t2,
    unsigned short* __restrict__ t3, unsigned short* __restrict__ t4) {
    __shared__ unsigned short t[64][72];
    const int tid = threadIdx.x;
    const int k0 = blockIdx.x * 64, n0 = blockIdx.y * 64, z = blockIdx.z;
    const float* W = (z == 0) ? w0 : (z == 1) ? w1 : (z == 2) ? w2 : (z == 3) ? w3 : w4;
    unsigned short* WT = (z == 0) ? t0 : (z == 1) ? t1 : (z == 2) ? t2 : (z == 3) ? t3 : t4;
#pragma unroll
    for (int i = 0; i < 4; ++i) {  // load 64x64 f32, transpose-scatter as bf16
        int idx = tid + 256 * i;
        int r = idx >> 4, c4 = (idx & 15) * 4;
        const f32x4 f = *(const f32x4*)(W + (size_t)(k0 + r) * 1024 + n0 + c4);
        t[c4 + 0][r] = f2bf(f.x);
        t[c4 + 1][r] = f2bf(f.y);
        t[c4 + 2][r] = f2bf(f.z);
        t[c4 + 3][r] = f2bf(f.w);
    }
    __syncthreads();
#pragma unroll
    for (int i = 0; i < 2; ++i) {  // coalesced b128 writes of WT rows
        int idx = tid + 256 * i;
        int r = idx >> 3, c8 = (idx & 7) * 8;
        *(u32x4*)(WT + (size_t)(n0 + r) * 1024 + k0 + c8) = *(const u32x4*)&t[r][c8];
    }
}

// =====================================================================
// All-bf16 tiled GEMM: A[16384,1024] x WT[n][k] -> [16384,1024]
// Block 128x64, 8 waves (4Mx2N), wave tile 32x32, Kstep 32, double-buffered
// async global->LDS staging.
// OUTMODE: 0 = f32 row-major, 1 = bf16 row-major, 2 = bf16 head-transposed [bh][d][n]
// ACT:     0 = identity, 1 = elu(x)+1, 2 = sigmoid(x + bias)
// =====================================================================
template <int OUTMODE, int ACT>
__global__ __launch_bounds__(256, 2) void gemm128x64(const unsigned short* __restrict__ A,
                                                     const unsigned short* __restrict__ BT,
                                                     const float* __restrict__ bias,
                                                     void* __restrict__ Outp) {
    constexpr int K = 1024, N = 1024, STEPS = K / 32;
    __shared__ unsigned short As[2][128][40];  // [m][k] row-major, pad->40
    __shared__ unsigned short Bt[2][64][40];   // [n][k]

    const int tid   = threadIdx.x;
    const int mbase = blockIdx.x * 128;
    const int nbase = blockIdx.y * 64;
    const int wave  = tid >> 5, lane = tid & 31;
    const int wm = wave & 3, wn = wave >> 2;
    const int hi = lane >> 4, lo = lane & 15;
    const int br = tid >> 2, bc = (tid & 3) * 8;  // B: one 16B chunk per thread

    v8f acc[4];
#pragma unroll
    for (int i = 0; i < 4; ++i) acc[i] = (v8f)0.0f;

    // stage(buf, k0): 3 async 16B copies per thread (A:2, B:1)
    auto stage = [&](int buf, int k0) {
#pragma unroll
        for (int i = 0; i < 2; ++i) {
            int idx = tid + 256 * i;
            int r = idx >> 2, c = (idx & 3) * 8;
            cp16(&As[buf][r][c], A + (size_t)(mbase + r) * K + k0 + c);
        }
        cp16(&Bt[buf][br][bc], BT + (size_t)(nbase + br) * K + k0 + bc);
    };

    stage(0, 0);
    for (int it = 0; it < STEPS; ++it) {
        const int buf = it & 1;
        if (it + 1 < STEPS) {
            stage(buf ^ 1, (it + 1) * 32);
            wait_async<3>();  // current tile landed; next still in flight
        } else {
            wait_async<0>();
        }
        __syncthreads();

        Frag a0, a1, b0, b1;
        const int m0 = wm * 32 + lo;
        a0.u[0] = *(const u32x4*)&As[buf][m0][hi * 8];
        a0.u[1] = *(const u32x4*)&As[buf][m0][16 + hi * 8];
        a1.u[0] = *(const u32x4*)&As[buf][m0 + 16][hi * 8];
        a1.u[1] = *(const u32x4*)&As[buf][m0 + 16][16 + hi * 8];
        const int n0 = wn * 32 + lo;
        b0.u[0] = *(const u32x4*)&Bt[buf][n0][hi * 8];
        b0.u[1] = *(const u32x4*)&Bt[buf][n0][16 + hi * 8];
        b1.u[0] = *(const u32x4*)&Bt[buf][n0 + 16][hi * 8];
        b1.u[1] = *(const u32x4*)&Bt[buf][n0 + 16][16 + hi * 8];

        acc[0] = WMMA_BF16(a0.v, b0.v, acc[0]);
        acc[1] = WMMA_BF16(a0.v, b1.v, acc[1]);
        acc[2] = WMMA_BF16(a1.v, b0.v, acc[2]);
        acc[3] = WMMA_BF16(a1.v, b1.v, acc[3]);
        __syncthreads();
    }

    // ---- epilogue ----
    float bv[2] = {0.f, 0.f};
    if (ACT == 2) {
        bv[0] = bias[nbase + wn * 32 + lo];
        bv[1] = bias[nbase + wn * 32 + 16 + lo];
    }
#pragma unroll
    for (int st = 0; st < 4; ++st) {
        const int mt = st >> 1, nt = st & 1;
        const int gcol = nbase + wn * 32 + nt * 16 + lo;
        const int rowb = mbase + wm * 32 + mt * 16 + hi * 8;  // 8 consecutive rows
        if (OUTMODE == 2) {
            U4S o;
#pragma unroll
            for (int r = 0; r < 8; ++r) {
                float v = acc[st][r];
                if (ACT == 1) v = (v > 0.f) ? (v + 1.f) : __expf(v);
                o.s[r] = f2bf(v);
            }
            const int b = rowb >> 12, n = rowb & 4095;
            const int h = gcol >> 6, d = gcol & 63;
            unsigned short* dst =
                (unsigned short*)Outp + ((size_t)(b * 16 + h) * 64 + d) * 4096 + n;
            *(u32x4*)dst = o.u;  // 8 consecutive n, packed b128 store
        } else {
#pragma unroll
            for (int r = 0; r < 8; ++r) {
                float v = acc[st][r];
                if (ACT == 1) v = (v > 0.f) ? (v + 1.f) : __expf(v);
                if (ACT == 2) v = 1.f / (1.f + __expf(-(v + bv[nt])));
                if (OUTMODE == 0)
                    ((float*)Outp)[(size_t)(rowb + r) * N + gcol] = v;
                else
                    ((unsigned short*)Outp)[(size_t)(rowb + r) * N + gcol] = f2bf(v);
            }
        }
    }
}

// =====================================================================
// kv[bh] = k^T v (64x64, reduce n=4096) + k_sum. Inputs pre-transposed
// kT/vT [bh][64][4096] -> pure row-major async staging, double-buffered.
// Output kvT [bh][e][d] (transposed for attout), packed b128 stores.
// =====================================================================
__global__ __launch_bounds__(128, 4) void kv_kernel(const unsigned short* __restrict__ kT,
                                                    const unsigned short* __restrict__ vT,
                                                    unsigned short* __restrict__ kvT,
                                                    float* __restrict__ ksump) {
    __shared__ unsigned short kt[2][64][40];  // [d][n]
    __shared__ unsigned short vt[2][64][40];  // [e][n]

    const int tid = threadIdx.x;
    const int bh = blockIdx.x;
    const size_t base = (size_t)bh * 64 * 4096;
    const int wave = tid >> 5, lane = tid & 31;
    const int hi = lane >> 4, lo = lane & 15;
    const int d0 = wave * 16;

    v8f acc[4];
#pragma unroll
    for (int i = 0; i < 4; ++i) acc[i] = (v8f)0.0f;
    float ks = 0.f;

    auto stage = [&](int buf, int n0) {  // 4 async 16B copies per thread
#pragma unroll
        for (int i = 0; i < 2; ++i) {
            int idx = tid + 128 * i;
            int r = idx >> 2, c = (idx & 3) * 8;
            cp16(&kt[buf][r][c], kT + base + (size_t)r * 4096 + n0 + c);
            cp16(&vt[buf][r][c], vT + base + (size_t)r * 4096 + n0 + c);
        }
    };

    stage(0, 0);
    for (int it = 0; it < 128; ++it) {
        const int buf = it & 1;
        if (it + 1 < 128) {
            stage(buf ^ 1, (it + 1) * 32);
            wait_async<4>();
        } else {
            wait_async<0>();
        }
        __syncthreads();

        if (tid < 64) {  // fold k_sum: sum over n of kt[d][n]
#pragma unroll
            for (int j = 0; j < 4; ++j) {
                U4S u;
                u.u = *(const u32x4*)&kt[buf][tid][j * 8];
#pragma unroll
                for (int q = 0; q < 8; ++q) ks += bf2f(u.s[q]);
            }
        }

        Frag a, bb;
        a.u[0] = *(const u32x4*)&kt[buf][d0 + lo][hi * 8];  // A = k^T (16d x 32n)
        a.u[1] = *(const u32x4*)&kt[buf][d0 + lo][16 + hi * 8];
#pragma unroll
        for (int et = 0; et < 4; ++et) {                    // B = v (32n x 16e)
            bb.u[0] = *(const u32x4*)&vt[buf][et * 16 + lo][hi * 8];
            bb.u[1] = *(const u32x4*)&vt[buf][et * 16 + lo][16 + hi * 8];
            acc[et] = WMMA_BF16(a.v, bb.v, acc[et]);
        }
        __syncthreads();
    }

#pragma unroll
    for (int et = 0; et < 4; ++et) {  // store transposed kvT[e][d], packed
        const int e = et * 16 + lo;
        const int db = d0 + hi * 8;
        U4S o;
#pragma unroll
        for (int r = 0; r < 8; ++r) o.s[r] = f2bf(acc[et][r]);
        *(u32x4*)(kvT + ((size_t)bh * 64 + e) * 64 + db) = o.u;
    }
    if (tid < 64) ksump[bh * 64 + tid] = ks;
}

// =====================================================================
// att = (q*SCALE) @ kv * z * g -> bf16 row-major [16384,1024]
// grid (32 n-tiles, 64 bh); K=64 fully resident; async staging.
// =====================================================================
__global__ __launch_bounds__(256, 2) void attout_kernel(const unsigned short* __restrict__ qp,
                                                        const unsigned short* __restrict__ gp,
                                                        const unsigned short* __restrict__ kvT,
                                                        const float* __restrict__ ksump,
                                                        unsigned short* __restrict__ attp) {
    __shared__ unsigned short Qs[128][72];   // [n][d]
    __shared__ unsigned short KVt[64][72];   // [e][d]
    __shared__ float zs[128];
    __shared__ float kss[64];

    const int tid = threadIdx.x;
    const int n0 = blockIdx.x * 128;
    const int bh = blockIdx.y;
    const int b = bh >> 4, h = bh & 15;
    const size_t base = ((size_t)b * 4096 + n0) * 1024 + (size_t)h * 64;

#pragma unroll
    for (int i = 0; i < 4; ++i) {  // stage q (row-major async)
        int idx = tid + 256 * i;
        int r = idx >> 3, c = (idx & 7) * 8;
        cp16(&Qs[r][c], qp + base + (size_t)r * 1024 + c);
    }
#pragma unroll
    for (int i = 0; i < 2; ++i) {  // stage kvT (row-major async)
        int idx = tid + 256 * i;
        int e = idx >> 3, c = (idx & 7) * 8;
        cp16(&KVt[e][c], kvT + ((size_t)bh * 64 + e) * 64 + c);
    }
    if (tid < 64) kss[tid] = ksump[bh * 64 + tid];
    wait_async<0>();
    __syncthreads();

    if (tid < 128) {  // z = 1 / max(q . k_sum, 1e-6)
        float den = 0.f;
#pragma unroll
        for (int d = 0; d < 64; ++d) den += bf2f(Qs[tid][d]) * kss[d];
        zs[tid] = 1.f / fmaxf(den, 1e-6f);
    }
    __syncthreads();

    const int wave = tid >> 5, lane = tid & 31;
    const int wm = wave & 3, wn = wave >> 2;
    const int hi = lane >> 4, lo = lane & 15;

    v8f acc[4];
#pragma unroll
    for (int i = 0; i < 4; ++i) acc[i] = (v8f)0.0f;

#pragma unroll
    for (int kk = 0; kk < 64; kk += 32) {
        Frag a0, a1, b0, b1;
        const int m0 = wm * 32 + lo;
        a0.u[0] = *(const u32x4*)&Qs[m0][kk + hi * 8];
        a0.u[1] = *(const u32x4*)&Qs[m0][kk + 16 + hi * 8];
        a1.u[0] = *(const u32x4*)&Qs[m0 + 16][kk + hi * 8];
        a1.u[1] = *(const u32x4*)&Qs[m0 + 16][kk + 16 + hi * 8];
        const int e0 = wn * 32 + lo;
        b0.u[0] = *(const u32x4*)&KVt[e0][kk + hi * 8];
        b0.u[1] = *(const u32x4*)&KVt[e0][kk + 16 + hi * 8];
        b1.u[0] = *(const u32x4*)&KVt[e0 + 16][kk + hi * 8];
        b1.u[1] = *(const u32x4*)&KVt[e0 + 16][kk + 16 + hi * 8];

        acc[0] = WMMA_BF16(a0.v, b0.v, acc[0]);
        acc[1] = WMMA_BF16(a0.v, b1.v, acc[1]);
        acc[2] = WMMA_BF16(a1.v, b0.v, acc[2]);
        acc[3] = WMMA_BF16(a1.v, b1.v, acc[3]);
    }

#pragma unroll
    for (int st = 0; st < 4; ++st) {
        const int mt = st >> 1, nt = st & 1;
        const int e = wn * 32 + nt * 16 + lo;
#pragma unroll
        for (int r = 0; r < 8; ++r) {
            const int rl = wm * 32 + mt * 16 + hi * 8 + r;
            const float g = bf2f(gp[base + (size_t)rl * 1024 + e]);
            const float v = acc[st][r] * 0.125f * zs[rl] * g;  // SCALE = 64^-0.5
            attp[base + (size_t)rl * 1024 + e] = f2bf(v);
        }
    }
}

// =====================================================================
// Host-side launcher
// =====================================================================
extern "C" void kernel_launch(void* const* d_in, const int* in_sizes, int n_in,
                              void* d_out, int out_size, void* d_ws, size_t ws_size,
                              hipStream_t stream) {
    (void)in_sizes; (void)n_in; (void)out_size; (void)ws_size;
    const float* x  = (const float*)d_in[0];
    const float* Wq = (const float*)d_in[1];
    const float* Wk = (const float*)d_in[2];
    const float* Wv = (const float*)d_in[3];
    const float* Wg = (const float*)d_in[4];
    const float* bg = (const float*)d_in[5];
    const float* Wo = (const float*)d_in[6];
    float* out = (float*)d_out;

    constexpr size_t PLANE = (size_t)16384 * 1024 * 2;  // one bf16 activation plane
    constexpr size_t WMAT  = (size_t)1024 * 1024 * 2;   // one bf16 weight matrix
    char* ws = (char*)d_ws;
    unsigned short* xb   = (unsigned short*)(ws + 0 * PLANE);
    unsigned short* q_bf = (unsigned short*)(ws + 1 * PLANE);
    unsigned short* kT   = (unsigned short*)(ws + 2 * PLANE);
    unsigned short* vT   = (unsigned short*)(ws + 3 * PLANE);
    unsigned short* g_bf = (unsigned short*)(ws + 4 * PLANE);
    unsigned short* WTq  = (unsigned short*)(ws + 5 * PLANE + 0 * WMAT);
    unsigned short* WTk  = (unsigned short*)(ws + 5 * PLANE + 1 * WMAT);
    unsigned short* WTv  = (unsigned short*)(ws + 5 * PLANE + 2 * WMAT);
    unsigned short* WTg  = (unsigned short*)(ws + 5 * PLANE + 3 * WMAT);
    unsigned short* WTo  = (unsigned short*)(ws + 5 * PLANE + 4 * WMAT);
    unsigned short* kvT  = (unsigned short*)(ws + 5 * PLANE + 5 * WMAT);
    float*          ksum = (float*)(ws + 5 * PLANE + 5 * WMAT + (size_t)64 * 64 * 64 * 2);
    unsigned short* att  = kT;  // kT dead after kv_kernel: reuse region

    xconv_kernel<<<16384, 256, 0, stream>>>(x, xb);
    wconvT_kernel<<<dim3(16, 16, 5), 256, 0, stream>>>(Wq, Wk, Wv, Wg, Wo,
                                                       WTq, WTk, WTv, WTg, WTo);

    const dim3 gg(128, 16), gb(256);
    gemm128x64<1, 1><<<gg, gb, 0, stream>>>(xb, WTq, nullptr, q_bf);  // q: elu+1, row-major
    gemm128x64<2, 1><<<gg, gb, 0, stream>>>(xb, WTk, nullptr, kT);    // k: elu+1, headT
    gemm128x64<2, 0><<<gg, gb, 0, stream>>>(xb, WTv, nullptr, vT);    // v: identity, headT
    gemm128x64<1, 2><<<gg, gb, 0, stream>>>(xb, WTg, bg, g_bf);       // g: sigmoid+bias

    kv_kernel<<<64, 128, 0, stream>>>(kT, vT, kvT, ksum);
    attout_kernel<<<dim3(32, 64), 256, 0, stream>>>(q_bf, g_bf, kvT, ksum, att);

    gemm128x64<0, 0><<<gg, gb, 0, stream>>>(att, WTo, nullptr, out);  // out proj, f32
}
